// Attention_59287728554369
// MI455X (gfx1250) — compile-verified, matrix operand order checked by reading, not run on hardware
//
#include <hip/hip_runtime.h>
#include <hip/hip_bf16.h>

typedef __attribute__((ext_vector_type(16))) __bf16 v16bf;
typedef __attribute__((ext_vector_type(8)))  __bf16 v8bf;
typedef __attribute__((ext_vector_type(4)))  __bf16 v4bf;
typedef __attribute__((ext_vector_type(8)))  float  v8f;

#define WMMA_BF16(a, b, c) \
  __builtin_amdgcn_wmma_f32_16x16x32_bf16(false, (a), false, (b), (short)0, (c), false, false)

// Build a 16-element bf16 A/B fragment from two contiguous 8-element (16B) runs.
__device__ __forceinline__ v16bf make_frag(const __bf16* p0, const __bf16* p1) {
  v8bf lo = *(const v8bf*)p0;
  v8bf hi = *(const v8bf*)p1;
  v16bf r;
#pragma unroll
  for (int i = 0; i < 8; ++i) { r[i] = lo[i]; r[i + 8] = hi[i]; }
  return r;
}

// ---- DPP16 butterfly reductions within each 16-lane half (no LDS traffic) ----
template <int CTRL>
__device__ __forceinline__ float dppf(float x) {
  return __int_as_float(
      __builtin_amdgcn_update_dpp(0, __float_as_int(x), CTRL, 0xf, 0xf, true));
}
__device__ __forceinline__ float rmax16(float x) {
  x = fmaxf(x, dppf<0xB1>(x));   // quad_perm xor1
  x = fmaxf(x, dppf<0x4E>(x));   // quad_perm xor2
  x = fmaxf(x, dppf<0x141>(x));  // row_half_mirror (xor4)
  x = fmaxf(x, dppf<0x140>(x));  // row_mirror (xor8)
  return x;
}
__device__ __forceinline__ float rsum16(float x) {
  x += dppf<0xB1>(x);
  x += dppf<0x4E>(x);
  x += dppf<0x141>(x);
  x += dppf<0x140>(x);
  return x;
}

// ---------------------------------------------------------------------------
// Generic K=1024 GEMM, block tile 128(M) x 64(N), K-step 64, 8 waves (4x2),
// each wave computes a 32x32 C tile with 4 bf16 WMMA accumulators.
// MODE 0: A=f32 q,   B=f32 Wq,  out -> qh bf16 [b,h,nq,64]  (x scale*log2e)
// MODE 1: A=f32 ctx, B=f32 Wkv, out -> kh/vh bf16 [b,h,nk,64]
// MODE 2: A=bf16 ao, B=f32 Wout (+bias), out -> f32 [m,1024]
// ---------------------------------------------------------------------------
template <int MODE>
__global__ __launch_bounds__(256) void gemm_k(
    const void* __restrict__ Ap, const float* __restrict__ Bw,
    const float* __restrict__ bias, __bf16* __restrict__ o0,
    __bf16* __restrict__ o1, float* __restrict__ of, int N) {
  constexpr int K = 1024;
  __shared__ __align__(16) __bf16 As[128][64];
  __shared__ __align__(16) __bf16 Bt[64][64];  // B stored transposed [n][k]

  const int tid = threadIdx.x;
  const int lane = tid & 31, wave = tid >> 5;
  const int wm = wave >> 1, wn = wave & 1;
  const int m0 = blockIdx.y * 128, n0 = blockIdx.x * 64;
  const int l16 = lane & 15, half = lane >> 4, kb = half * 8;

  v8f acc[4] = {v8f{}, v8f{}, v8f{}, v8f{}};

  for (int kk = 0; kk < K; kk += 64) {
    if (MODE == 2) {
      const __bf16* A = (const __bf16*)Ap;
#pragma unroll
      for (int i = 0; i < 4; ++i) {
        int idx = tid + i * 256;
        int row = idx >> 3, seg = (idx & 7) * 8;
        *(v8bf*)&As[row][seg] = *(const v8bf*)&A[(size_t)(m0 + row) * K + kk + seg];
      }
    } else {
      const float* A = (const float*)Ap;
#pragma unroll
      for (int i = 0; i < 8; ++i) {
        int idx = tid + i * 256;
        int row = idx >> 4, seg = (idx & 15) * 4;
        const float4 v = *(const float4*)&A[(size_t)(m0 + row) * K + kk + seg];
        v4bf b;
        b[0] = (__bf16)v.x; b[1] = (__bf16)v.y; b[2] = (__bf16)v.z; b[3] = (__bf16)v.w;
        *(v4bf*)&As[row][seg] = b;
      }
    }
#pragma unroll
    for (int i = 0; i < 4; ++i) {
      int idx = tid + i * 256;
      int kr = idx >> 4, nq = (idx & 15) * 4;
      const float4 v = *(const float4*)&Bw[(size_t)(kk + kr) * N + n0 + nq];
      Bt[nq + 0][kr] = (__bf16)v.x;
      Bt[nq + 1][kr] = (__bf16)v.y;
      Bt[nq + 2][kr] = (__bf16)v.z;
      Bt[nq + 3][kr] = (__bf16)v.w;
    }
    __syncthreads();

#pragma unroll
    for (int kc = 0; kc < 64; kc += 32) {
      v16bf a0 = make_frag(&As[wm * 32 + l16][kc + kb],      &As[wm * 32 + l16][kc + kb + 16]);
      v16bf a1 = make_frag(&As[wm * 32 + 16 + l16][kc + kb], &As[wm * 32 + 16 + l16][kc + kb + 16]);
      v16bf b0 = make_frag(&Bt[wn * 32 + l16][kc + kb],      &Bt[wn * 32 + l16][kc + kb + 16]);
      v16bf b1 = make_frag(&Bt[wn * 32 + 16 + l16][kc + kb], &Bt[wn * 32 + 16 + l16][kc + kb + 16]);
      acc[0] = WMMA_BF16(a0, b0, acc[0]);
      acc[1] = WMMA_BF16(a0, b1, acc[1]);
      acc[2] = WMMA_BF16(a1, b0, acc[2]);
      acc[3] = WMMA_BF16(a1, b1, acc[3]);
    }
    __syncthreads();
  }

  // fold attention scale (in log2 domain) into Q so flash uses raw exp2
  constexpr float osc = (MODE == 0) ? 0.18033688011112042f /* 0.125*log2(e) */ : 1.0f;

#pragma unroll
  for (int am = 0; am < 2; ++am)
#pragma unroll
    for (int bn = 0; bn < 2; ++bn) {
      const v8f c = acc[am * 2 + bn];
#pragma unroll
      for (int r = 0; r < 8; ++r) {
        int m = m0 + wm * 32 + am * 16 + r + 8 * half;
        int n = n0 + wn * 32 + bn * 16 + l16;
        float val = c[r] * osc;
        if (MODE == 2) {
          of[(size_t)m * N + n] = val + bias[n];
        } else {
          int bb = m >> 11, ri = m & 2047;
          int cc = n;
          __bf16* dst = o0;
          if (MODE == 1 && cc >= 1024) { dst = o1; cc -= 1024; }
          int h = cc >> 6, d = cc & 63;
          dst[(((size_t)(bb * 16 + h)) * 2048 + ri) * 64 + d] = (__bf16)val;
        }
      }
    }
}

// ---------------------------------------------------------------------------
// Flash attention: grid (nq/128, b*h). 8 waves, each owns 16 q rows; stream
// over 64-key tiles with online softmax in the log2 domain (scale pre-folded
// into Q). dh = 64.
// ---------------------------------------------------------------------------
__global__ __launch_bounds__(256) void flash_k(
    const __bf16* __restrict__ qh, const __bf16* __restrict__ kh,
    const __bf16* __restrict__ vh, const unsigned char* __restrict__ mask,
    __bf16* __restrict__ ao) {
  __shared__ __align__(16) __bf16 Ks[64][64];      // [j][d]
  __shared__ __align__(16) __bf16 Vt[64][64];      // [d][j] (transposed)
  __shared__ __align__(16) __bf16 Ps[8][16][64];   // per-wave P round-trip

  const int tid = threadIdx.x, lane = tid & 31, w = tid >> 5;
  const int l16 = lane & 15, half = lane >> 4, kb = half * 8;
  const int bh = blockIdx.y;               // bb*16 + hh
  const int bb = bh >> 4, hh = bh & 15;
  const int qr = blockIdx.x * 128 + w * 16;

  // Q fragments for this wave's 16 rows, kept in registers for the whole pass.
  const __bf16* qbase = qh + ((size_t)bh * 2048 + qr + l16) * 64;
  const v16bf qa0 = make_frag(qbase + kb,      qbase + kb + 16);   // d = 0..31
  const v16bf qa1 = make_frag(qbase + 32 + kb, qbase + 48 + kb);   // d = 32..63

  v8f o[4] = {v8f{}, v8f{}, v8f{}, v8f{}};
  float mrow[8], lrow[8];
#pragma unroll
  for (int r = 0; r < 8; ++r) { mrow[r] = -3.0e38f; lrow[r] = 0.0f; }

  for (int kt = 0; kt < 2048; kt += 64) {
    // stage K (row major) and V (transposed) 64x64 tiles
#pragma unroll
    for (int i = 0; i < 2; ++i) {
      int idx = tid + i * 256;
      int j = idx >> 3, dq = (idx & 7) * 8;
      const size_t base = ((size_t)bh * 2048 + kt + j) * 64 + dq;
      *(v8bf*)&Ks[j][dq] = *(const v8bf*)&kh[base];
      v8bf vv = *(const v8bf*)&vh[base];
#pragma unroll
      for (int c = 0; c < 8; ++c) Vt[dq + c][j] = vv[c];
    }
    __syncthreads();

    float mb[4];
#pragma unroll
    for (int jt = 0; jt < 4; ++jt)
      mb[jt] = mask[(size_t)bb * 2048 + kt + jt * 16 + l16] ? -1.0e30f : 0.0f;

    // S = Q K^T : four 16x16 tiles over 64 keys (scores already in log2 units)
    v8f s[4];
#pragma unroll
    for (int jt = 0; jt < 4; ++jt) {
      v16bf ka = make_frag(&Ks[jt * 16 + l16][kb],      &Ks[jt * 16 + l16][kb + 16]);
      v16bf kc = make_frag(&Ks[jt * 16 + l16][32 + kb], &Ks[jt * 16 + l16][48 + kb]);
      v8f t = v8f{};
      t = WMMA_BF16(qa0, ka, t);
      t = WMMA_BF16(qa1, kc, t);
      s[jt] = t;
    }

    // Online softmax update (row r+8*half lives in this lane's half).
#pragma unroll
    for (int r = 0; r < 8; ++r) {
      float a0 = s[0][r] + mb[0];
      float a1 = s[1][r] + mb[1];
      float a2 = s[2][r] + mb[2];
      float a3 = s[3][r] + mb[3];
      float tm = rmax16(fmaxf(fmaxf(a0, a1), fmaxf(a2, a3)));
      float nm = fmaxf(mrow[r], tm);
      float f = __builtin_amdgcn_exp2f(mrow[r] - nm);
      mrow[r] = nm;
      float p0 = __builtin_amdgcn_exp2f(a0 - nm);
      float p1 = __builtin_amdgcn_exp2f(a1 - nm);
      float p2 = __builtin_amdgcn_exp2f(a2 - nm);
      float p3 = __builtin_amdgcn_exp2f(a3 - nm);
      s[0][r] = p0; s[1][r] = p1; s[2][r] = p2; s[3][r] = p3;
      float ps = rsum16((p0 + p1) + (p2 + p3));
      lrow[r] = lrow[r] * f + ps;
      o[0][r] *= f; o[1][r] *= f; o[2][r] *= f; o[3][r] *= f;
    }

    // P (f32 C-layout) -> bf16 A-layout via wave-private LDS round-trip.
#pragma unroll
    for (int r = 0; r < 8; ++r) {
#pragma unroll
      for (int jt = 0; jt < 4; ++jt)
        Ps[w][r + 8 * half][jt * 16 + l16] = (__bf16)s[jt][r];
    }
    asm volatile("s_wait_dscnt 0" ::: "memory");
    v16bf pa0 = make_frag(&Ps[w][l16][kb],      &Ps[w][l16][kb + 16]);
    v16bf pa1 = make_frag(&Ps[w][l16][32 + kb], &Ps[w][l16][48 + kb]);

    // O += P * V  (4 column tiles of dh, contraction over 64 keys)
#pragma unroll
    for (int dt = 0; dt < 4; ++dt) {
      v16bf vb0 = make_frag(&Vt[dt * 16 + l16][kb],      &Vt[dt * 16 + l16][kb + 16]);
      v16bf vb1 = make_frag(&Vt[dt * 16 + l16][32 + kb], &Vt[dt * 16 + l16][48 + kb]);
      o[dt] = WMMA_BF16(pa0, vb0, o[dt]);
      o[dt] = WMMA_BF16(pa1, vb1, o[dt]);
    }
    __syncthreads();
  }

  // normalize and write to [b, nq, h*64] bf16
#pragma unroll
  for (int r = 0; r < 8; ++r) {
    float rinv = 1.0f / lrow[r];
    int qrow = qr + r + 8 * half;
    __bf16* dst = ao + ((size_t)bb * 2048 + qrow) * 1024 + hh * 64;
#pragma unroll
    for (int dt = 0; dt < 4; ++dt) dst[dt * 16 + l16] = (__bf16)(o[dt][r] * rinv);
  }
}

extern "C" void kernel_launch(void* const* d_in, const int* in_sizes, int n_in,
                              void* d_out, int out_size, void* d_ws, size_t ws_size,
                              hipStream_t stream) {
  const float* q    = (const float*)d_in[0];   // [2,2048,1024]
  const float* ctx  = (const float*)d_in[1];   // [2,2048,1024]
  const unsigned char* mask = (const unsigned char*)d_in[2];  // [2,2048] bool
  const float* Wq   = (const float*)d_in[3];   // [1024,1024]
  const float* Wkv  = (const float*)d_in[4];   // [1024,2048]
  const float* Wout = (const float*)d_in[5];   // [1024,1024]
  const float* bo   = (const float*)d_in[6];   // [1024]
  float* out = (float*)d_out;                  // [2,2048,1024]

  const size_t HSZ = (size_t)2 * 16 * 2048 * 64;  // 4 Mi elements per tensor
  __bf16* qh = (__bf16*)d_ws;
  __bf16* kh = qh + HSZ;
  __bf16* vh = kh + HSZ;
  __bf16* ao = vh + HSZ;  // [2,2048,1024] bf16

  // 1) Q projection (scaled) -> qh [b,h,nq,64]
  gemm_k<0><<<dim3(16, 32), 256, 0, stream>>>(q, Wq, nullptr, qh, nullptr, nullptr, 1024);
  // 2) KV projection -> kh/vh [b,h,nk,64]
  gemm_k<1><<<dim3(32, 32), 256, 0, stream>>>(ctx, Wkv, nullptr, kh, vh, nullptr, 2048);
  // 3) Fused flash attention -> ao [b,nq,h*64]
  flash_k<<<dim3(16, 32), 256, 0, stream>>>(qh, kh, vh, mask, ao);
  // 4) Output projection + bias -> d_out
  gemm_k<2><<<dim3(16, 32), 256, 0, stream>>>(ao, Wout, bo, nullptr, nullptr, out, 1024);
}
